// ParallelMHA_83056077570740
// MI455X (gfx1250) — matrix operand order repaired and checked by reading, not compiled, not run on hardware
//
#include <hip/hip_runtime.h>
#include <hip/hip_bf16.h>

// ---------------------------------------------------------------------------
// Causal MHA for MI455X (gfx1250), bf16 WMMA pipeline:
//   cast f32->bf16 -> fused QKV GEMM -> flash attention -> O-proj GEMM (+bias)
// GEMM tiles staged via async global->LDS copies (ASYNCcnt) with LDS double
// buffering when the toolchain exposes the gfx1250 async builtins.
// ---------------------------------------------------------------------------

#define BB   2
#define SEQ  2048
#define DIM  1024
#define NH   16
#define HDIM 64
#define MTOT (BB * SEQ) // 4096

typedef __bf16 bf16;
typedef __attribute__((ext_vector_type(16))) __bf16 v16bf;
typedef __attribute__((ext_vector_type(8)))  float  v8f;

// the async-load builtin takes int4 (gcc vector) pointers: AS1 src, AS3 dst
typedef int vint4 __attribute__((vector_size(16)));
typedef __attribute__((address_space(1))) vint4 g_vint4;
typedef __attribute__((address_space(3))) vint4 l_vint4;

#if defined(__gfx1250__) &&                                            \
    __has_builtin(__builtin_amdgcn_global_load_async_to_lds_b128) &&   \
    __has_builtin(__builtin_amdgcn_s_wait_asynccnt)
#define USE_ASYNC_COPY 1
#else
#define USE_ASYNC_COPY 0
#endif

__device__ __forceinline__ v8f wmma_bf16(v16bf a, v16bf b, v8f c) {
  // D = A(16x32 bf16) x B(32x16 bf16) + C(16x16 f32)
  return __builtin_amdgcn_wmma_f32_16x16x32_bf16(
      /*neg_a=*/false, a, /*neg_b=*/false, b,
      /*c_mod=*/(short)0, c, /*reuse_a=*/false, /*reuse_b=*/false);
}

// A-fragment viewed as 8 dwords: dword g holds bf16 pair (K, K+1) with
// K = g*2 + half*16   (contiguous k order, matching B fragments)
union AFrag {
  v16bf v;
  unsigned u[8];
};

// Stage 32 bytes (16 bf16) from global to LDS: async path uses two
// global_load_async_to_lds_b128 (ASYNCcnt), fallback is a plain v16bf copy.
__device__ __forceinline__ void stage32(const bf16* sp, bf16* lp) {
#if USE_ASYNC_COPY
  g_vint4* gp = (g_vint4*)sp;
  l_vint4* lq = (l_vint4*)lp;
  __builtin_amdgcn_global_load_async_to_lds_b128(gp,     lq,     /*off=*/0,
                                                 /*cpol=*/0);
  __builtin_amdgcn_global_load_async_to_lds_b128(gp + 1, lq + 1, /*off=*/0,
                                                 /*cpol=*/0);
#else
  *(v16bf*)lp = *(const v16bf*)sp;
  __builtin_prefetch(sp + 512, 0, 1); // pull next rows toward the WGP
#endif
}

__device__ __forceinline__ void wait_stage() {
#if USE_ASYNC_COPY
  __builtin_amdgcn_s_wait_asynccnt(0);
#endif
}

// ---------------------------------------------------------------------------
// Kernel 1: f32 -> bf16 cast (n must be a multiple of 4; all our sizes are)
// ---------------------------------------------------------------------------
__global__ void cast_f32_to_bf16(const float* __restrict__ src,
                                 bf16* __restrict__ dst, int n) {
  int i = (blockIdx.x * blockDim.x + threadIdx.x) * 4;
  if (i < n) {
    float4 f = *(const float4*)(src + i);
    dst[i + 0] = (bf16)f.x;
    dst[i + 1] = (bf16)f.y;
    dst[i + 2] = (bf16)f.z;
    dst[i + 3] = (bf16)f.w;
  }
}

// ---------------------------------------------------------------------------
// Kernel 2: fused QKV projection.  Y = X @ W.T for W in {Wq,Wk,Wv}.
// Block = 128 threads (4 waves), 64x64 output tile, each wave owns a 32x32
// quadrant (2x2 WMMA tiles) for all three outputs.  X tile double-buffered in
// LDS (async copy of tile k+1 overlaps WMMAs on tile k).  B-fragments read
// weight rows straight from global (row-major W == B layout, L2-resident).
// Q,K written as [B,H,N,HD]; V written transposed as [B,H,HD,N].
// ---------------------------------------------------------------------------
__global__ __launch_bounds__(128)
void qkv_gemm_kernel(const bf16* __restrict__ X,
                     const bf16* __restrict__ Wq,
                     const bf16* __restrict__ Wk,
                     const bf16* __restrict__ Wv,
                     bf16* __restrict__ Qo,
                     bf16* __restrict__ Ko,
                     bf16* __restrict__ Vt) {
  __shared__ bf16 xs[2][64][40]; // 2 x (64 rows x 32 cols), 16B-aligned rows

  const int tid  = threadIdx.x;
  const int wave = tid >> 5;
  const int lane = tid & 31;
  const int ln   = lane & 15;
  const int hf   = lane >> 4;
  const int wr   = wave >> 1; // 0..1
  const int wc   = wave & 1;  // 0..1
  const int m_blk = blockIdx.x * 64;
  const int n_blk = blockIdx.y * 64;
  const int m0 = m_blk + wr * 32;
  const int n0 = n_blk + wc * 32;

  const bf16* __restrict__ Wp[3] = {Wq, Wk, Wv};

  v8f acc[3][2][2];
#pragma unroll
  for (int w = 0; w < 3; ++w)
#pragma unroll
    for (int mt = 0; mt < 2; ++mt)
#pragma unroll
      for (int nt = 0; nt < 2; ++nt) acc[w][mt][nt] = (v8f){};

  const int stg_row = tid >> 1;
  const int stg_kh  = (tid & 1) * 16;
  const bf16* stg_src = X + (size_t)(m_blk + stg_row) * DIM + stg_kh;

  // prologue: stage tile 0 into buffer 0
  stage32(stg_src, &xs[0][stg_row][stg_kh]);

  for (int kc = 0; kc < DIM / 32; ++kc) {
    wait_stage();      // our async writes for tile kc have landed in LDS
    __syncthreads();   // everyone's writes visible; prior reads of buf done

    if (kc + 1 < DIM / 32)
      stage32(stg_src + (kc + 1) * 32, &xs[(kc + 1) & 1][stg_row][stg_kh]);

    // ---- A fragments for this wave's two 16-row subtiles ----
    AFrag a[2];
#pragma unroll
    for (int mt = 0; mt < 2; ++mt) {
      const int row = wr * 32 + mt * 16 + ln;
#pragma unroll
      for (int g = 0; g < 8; ++g) {
        const int pb = g * 2 + hf * 16;
        a[mt].u[g] = *(const unsigned*)(&xs[kc & 1][row][pb]);
      }
    }

    // ---- B fragments straight from global weight rows + WMMA ----
#pragma unroll
    for (int w = 0; w < 3; ++w) {
#pragma unroll
      for (int nt = 0; nt < 2; ++nt) {
        const bf16* bp =
            Wp[w] + (size_t)(n0 + nt * 16 + ln) * DIM + kc * 32 + hf * 16;
        v16bf bfr = *(const v16bf*)bp;
        acc[w][0][nt] = wmma_bf16(a[0].v, bfr, acc[w][0][nt]);
        acc[w][1][nt] = wmma_bf16(a[1].v, bfr, acc[w][1][nt]);
      }
    }
  }

  // ---- writeback: Q/K -> [B,H,N,HD], V -> [B,H,HD,N] (transposed) ----
#pragma unroll
  for (int w = 0; w < 3; ++w)
#pragma unroll
    for (int mt = 0; mt < 2; ++mt)
#pragma unroll
      for (int nt = 0; nt < 2; ++nt)
#pragma unroll
        for (int g = 0; g < 8; ++g) {
          const int m = m0 + mt * 16 + g + 8 * hf; // global row in [0,4096)
          const int n = n0 + nt * 16 + ln;         // global col in [0,1024)
          const int bb = m >> 11, s = m & (SEQ - 1);
          const int hh = n >> 6,  d = n & (HDIM - 1);
          const bf16 val = (bf16)acc[w][mt][nt][g];
          if (w == 0)
            Qo[((size_t)(bb * NH + hh) * SEQ + s) * HDIM + d] = val;
          else if (w == 1)
            Ko[((size_t)(bb * NH + hh) * SEQ + s) * HDIM + d] = val;
          else
            Vt[((size_t)(bb * NH + hh) * HDIM + d) * SEQ + s] = val;
        }
}

// ---------------------------------------------------------------------------
// Kernel 3: causal flash attention.  1 wave per (b,h, 16-query tile).
// 32 keys per iteration: 4 WMMAs for S = Q K^T, online softmax with 16-lane
// butterfly reductions, P re-swizzled via LDS, 4 WMMAs for O += P V.
// ---------------------------------------------------------------------------
__global__ __launch_bounds__(32)
void attn_kernel(const bf16* __restrict__ Q,
                 const bf16* __restrict__ K,
                 const bf16* __restrict__ Vt,
                 bf16* __restrict__ O) {
  __shared__ bf16 ps[16][40]; // P tile 16x32, padded

  const int lane = threadIdx.x & 31;
  const int ln = lane & 15, hf = lane >> 4;
  const int qt = blockIdx.x;
  const int bh = blockIdx.y; // b*NH + h
  const int bb = bh >> 4, hh = bh & 15;
  const int q0 = qt * 16;

  const bf16* __restrict__ Qb = Q  + (size_t)bh * SEQ * HDIM;
  const bf16* __restrict__ Kb = K  + (size_t)bh * SEQ * HDIM;
  const bf16* __restrict__ Vb = Vt + (size_t)bh * HDIM * SEQ;

  // Q A-fragments (HD=64 -> two 16x32 slabs)
  AFrag aq[2];
  {
    const bf16* qrow = Qb + (size_t)(q0 + ln) * HDIM;
#pragma unroll
    for (int kc = 0; kc < 2; ++kc)
#pragma unroll
      for (int g = 0; g < 8; ++g) {
        const int pb = kc * 32 + g * 2 + hf * 16;
        aq[kc].u[g] = *(const unsigned*)(qrow + pb);
      }
  }

  v8f o0 = (v8f){}, o1 = (v8f){}, o2 = (v8f){}, o3 = (v8f){};
  float mrow[8], lrow[8];
#pragma unroll
  for (int g = 0; g < 8; ++g) {
    mrow[g] = -__builtin_inff();
    lrow[g] = 0.f;
  }

  for (int j = 0; j < q0 + 16; j += 32) {
    // ---- S = (Q K^T) * 1/sqrt(HD), two 16x16 column tiles ----
    v8f s0 = (v8f){}, s1 = (v8f){};
#pragma unroll
    for (int kc = 0; kc < 2; ++kc) {
      v16bf b0 = *(const v16bf*)(Kb + (size_t)(j + ln) * HDIM + kc * 32 + hf * 16);
      v16bf b1 = *(const v16bf*)(Kb + (size_t)(j + 16 + ln) * HDIM + kc * 32 + hf * 16);
      s0 = wmma_bf16(aq[kc].v, b0, s0);
      s1 = wmma_bf16(aq[kc].v, b1, s1);
    }

    const int qbase = q0 + 8 * hf;
#pragma unroll
    for (int g = 0; g < 8; ++g) {
      const int qi = qbase + g; // query row of this element
      float v0 = s0[g] * 0.125f;
      float v1 = s1[g] * 0.125f;
      if (j + ln > qi)      v0 = -__builtin_inff(); // causal mask
      if (j + 16 + ln > qi) v1 = -__builtin_inff();
      s0[g] = v0;
      s1[g] = v1;
    }

    // ---- online softmax (row = one 16-lane half-group, element g) ----
    float corr[8];
#pragma unroll
    for (int g = 0; g < 8; ++g) {
      float mx = fmaxf(s0[g], s1[g]);
      mx = fmaxf(mx, __shfl_xor(mx, 1, 32));
      mx = fmaxf(mx, __shfl_xor(mx, 2, 32));
      mx = fmaxf(mx, __shfl_xor(mx, 4, 32));
      mx = fmaxf(mx, __shfl_xor(mx, 8, 32));
      const float mnew = fmaxf(mrow[g], mx);
      const float p0 = __expf(s0[g] - mnew);
      const float p1 = __expf(s1[g] - mnew);
      float rs = p0 + p1;
      rs += __shfl_xor(rs, 1, 32);
      rs += __shfl_xor(rs, 2, 32);
      rs += __shfl_xor(rs, 4, 32);
      rs += __shfl_xor(rs, 8, 32);
      const float c = __expf(mrow[g] - mnew);
      corr[g] = c;
      lrow[g] = lrow[g] * c + rs;
      mrow[g] = mnew;
      s0[g] = p0;
      s1[g] = p1;
    }
#pragma unroll
    for (int g = 0; g < 8; ++g) {
      o0[g] *= corr[g]; o1[g] *= corr[g];
      o2[g] *= corr[g]; o3[g] *= corr[g];
    }

    // ---- P (16x32) through LDS to A-fragment layout ----
#pragma unroll
    for (int g = 0; g < 8; ++g) {
      ps[g + 8 * hf][ln]      = (bf16)s0[g];
      ps[g + 8 * hf][16 + ln] = (bf16)s1[g];
    }
    __syncthreads();
    AFrag ap;
#pragma unroll
    for (int g = 0; g < 8; ++g) {
      const int pb = g * 2 + hf * 16;
      ap.u[g] = *(const unsigned*)(&ps[ln][pb]);
    }
    __syncthreads();

    // ---- O += P V  (B-frags = contiguous rows of transposed V) ----
    v16bf bv;
    bv = *(const v16bf*)(Vb + (size_t)(0 * 16 + ln) * SEQ + j + hf * 16);
    o0 = wmma_bf16(ap.v, bv, o0);
    bv = *(const v16bf*)(Vb + (size_t)(1 * 16 + ln) * SEQ + j + hf * 16);
    o1 = wmma_bf16(ap.v, bv, o1);
    bv = *(const v16bf*)(Vb + (size_t)(2 * 16 + ln) * SEQ + j + hf * 16);
    o2 = wmma_bf16(ap.v, bv, o2);
    bv = *(const v16bf*)(Vb + (size_t)(3 * 16 + ln) * SEQ + j + hf * 16);
    o3 = wmma_bf16(ap.v, bv, o3);
  }

  // ---- finalize: O /= l, write head-concat [B,N,D] bf16 ----
#pragma unroll
  for (int g = 0; g < 8; ++g) {
    const float inv = 1.f / lrow[g];
    const int s = q0 + g + 8 * hf;
    const size_t rowoff = ((size_t)bb * SEQ + s) * DIM + hh * HDIM;
    O[rowoff + 0 * 16 + ln] = (bf16)(o0[g] * inv);
    O[rowoff + 1 * 16 + ln] = (bf16)(o1[g] * inv);
    O[rowoff + 2 * 16 + ln] = (bf16)(o2[g] * inv);
    O[rowoff + 3 * 16 + ln] = (bf16)(o3[g] * inv);
  }
}

// ---------------------------------------------------------------------------
// Kernel 4: output projection  out = A @ Wo.T + bo   (f32 out)
// Same tiling as QKV kernel (double-buffered async LDS staging).
// ---------------------------------------------------------------------------
__global__ __launch_bounds__(128)
void oproj_gemm_kernel(const bf16* __restrict__ X,
                       const bf16* __restrict__ W,
                       const float* __restrict__ bias,
                       float* __restrict__ out) {
  __shared__ bf16 xs[2][64][40];

  const int tid  = threadIdx.x;
  const int wave = tid >> 5;
  const int lane = tid & 31;
  const int ln   = lane & 15;
  const int hf   = lane >> 4;
  const int wr   = wave >> 1;
  const int wc   = wave & 1;
  const int m_blk = blockIdx.x * 64;
  const int n_blk = blockIdx.y * 64;
  const int m0 = m_blk + wr * 32;
  const int n0 = n_blk + wc * 32;

  v8f acc[2][2];
#pragma unroll
  for (int mt = 0; mt < 2; ++mt)
#pragma unroll
    for (int nt = 0; nt < 2; ++nt) acc[mt][nt] = (v8f){};

  const int stg_row = tid >> 1;
  const int stg_kh  = (tid & 1) * 16;
  const bf16* stg_src = X + (size_t)(m_blk + stg_row) * DIM + stg_kh;

  stage32(stg_src, &xs[0][stg_row][stg_kh]);

  for (int kc = 0; kc < DIM / 32; ++kc) {
    wait_stage();
    __syncthreads();

    if (kc + 1 < DIM / 32)
      stage32(stg_src + (kc + 1) * 32, &xs[(kc + 1) & 1][stg_row][stg_kh]);

    AFrag a[2];
#pragma unroll
    for (int mt = 0; mt < 2; ++mt) {
      const int row = wr * 32 + mt * 16 + ln;
#pragma unroll
      for (int g = 0; g < 8; ++g) {
        const int pb = g * 2 + hf * 16;
        a[mt].u[g] = *(const unsigned*)(&xs[kc & 1][row][pb]);
      }
    }
#pragma unroll
    for (int nt = 0; nt < 2; ++nt) {
      const bf16* bp = W + (size_t)(n0 + nt * 16 + ln) * DIM + kc * 32 + hf * 16;
      v16bf bfr = *(const v16bf*)bp;
      acc[0][nt] = wmma_bf16(a[0].v, bfr, acc[0][nt]);
      acc[1][nt] = wmma_bf16(a[1].v, bfr, acc[1][nt]);
    }
  }

#pragma unroll
  for (int mt = 0; mt < 2; ++mt)
#pragma unroll
    for (int nt = 0; nt < 2; ++nt)
#pragma unroll
      for (int g = 0; g < 8; ++g) {
        const int m = m0 + mt * 16 + g + 8 * hf;
        const int n = n0 + nt * 16 + ln;
        out[(size_t)m * DIM + n] = acc[mt][nt][g] + bias[n];
      }
}

// ---------------------------------------------------------------------------
// Host launcher
// ---------------------------------------------------------------------------
extern "C" void kernel_launch(void* const* d_in, const int* in_sizes, int n_in,
                              void* d_out, int out_size, void* d_ws,
                              size_t ws_size, hipStream_t stream) {
  const float* x  = (const float*)d_in[0];
  const float* Wq = (const float*)d_in[1];
  const float* Wk = (const float*)d_in[2];
  const float* Wv = (const float*)d_in[3];
  const float* Wo = (const float*)d_in[4];
  const float* bo = (const float*)d_in[5];
  float* out = (float*)d_out;

  char* ws = (char*)d_ws;
  // workspace layout (bf16 buffers), 48 MB total
  bf16* Xbf = (bf16*)(ws + (size_t)0);           // 4096x1024      = 8 MB
  bf16* Wqb = (bf16*)(ws + ((size_t)8  << 20));  // 1024x1024      = 2 MB
  bf16* Wkb = (bf16*)(ws + ((size_t)10 << 20));
  bf16* Wvb = (bf16*)(ws + ((size_t)12 << 20));
  bf16* Wob = (bf16*)(ws + ((size_t)14 << 20));
  bf16* Qb  = (bf16*)(ws + ((size_t)16 << 20));  // [B,H,N,HD]     = 8 MB
  bf16* Kb  = (bf16*)(ws + ((size_t)24 << 20));  // [B,H,N,HD]     = 8 MB
  bf16* Vtb = (bf16*)(ws + ((size_t)32 << 20));  // [B,H,HD,N]     = 8 MB
  bf16* Ab  = (bf16*)(ws + ((size_t)40 << 20));  // [B,N,D]        = 8 MB

  const int nX = MTOT * DIM;   // 4,194,304
  const int nW = DIM * DIM;    // 1,048,576

  cast_f32_to_bf16<<<nX / (256 * 4), 256, 0, stream>>>(x, Xbf, nX);
  cast_f32_to_bf16<<<nW / (256 * 4), 256, 0, stream>>>(Wq, Wqb, nW);
  cast_f32_to_bf16<<<nW / (256 * 4), 256, 0, stream>>>(Wk, Wkb, nW);
  cast_f32_to_bf16<<<nW / (256 * 4), 256, 0, stream>>>(Wv, Wvb, nW);
  cast_f32_to_bf16<<<nW / (256 * 4), 256, 0, stream>>>(Wo, Wob, nW);

  qkv_gemm_kernel<<<dim3(MTOT / 64, DIM / 64), 128, 0, stream>>>(
      Xbf, Wqb, Wkb, Wvb, Qb, Kb, Vtb);

  attn_kernel<<<dim3(SEQ / 16, BB * NH), 32, 0, stream>>>(Qb, Kb, Vtb, Ab);

  oproj_gemm_kernel<<<dim3(MTOT / 64, DIM / 64), 128, 0, stream>>>(
      Ab, Wob, bo, out);
}
